// MoEAudioProjector_31610959298941
// MI455X (gfx1250) — matrix operand order repaired and checked by reading, not compiled, not run on hardware
//
#include <hip/hip_runtime.h>

typedef __bf16 bf16_t;
typedef __attribute__((ext_vector_type(8)))  bf16_t v8bf;
typedef __attribute__((ext_vector_type(16))) bf16_t v16bf;
typedef __attribute__((ext_vector_type(8)))  float  v8f;
typedef __attribute__((ext_vector_type(4)))  unsigned int u32x4;
typedef __attribute__((ext_vector_type(4)))  int        i32x4;

#if defined(__HIP_DEVICE_COMPILE__) && defined(__has_builtin)
#if __has_builtin(__builtin_amdgcn_global_load_async_to_lds_b128)
#define HAVE_ASYNC_LDS 1
#endif
#if __has_builtin(__builtin_amdgcn_s_wait_asynccnt)
#define HAVE_WAIT_ASYNC_BUILTIN 1
#endif
#endif

#if defined(HAVE_ASYNC_LDS)
typedef __attribute__((address_space(1))) i32x4 gvec_t;  // global int4
typedef __attribute__((address_space(3))) i32x4 lvec_t;  // LDS int4
#endif

namespace {
constexpr int kInDim = 5120;   // ENC_DIM * K
constexpr int kHid   = 2048;
constexpr int kOut   = 2048;
constexpr int kExp   = 4;
constexpr int kTok   = 6000;   // 16 * 375
constexpr int BM = 128, BN = 128, BK = 64;
constexpr int kTileA = BM * BK;   // elements per A buffer
constexpr int kTileB = BN * BK;
constexpr float kEpsNorm = 1e-6f;
constexpr float kEpsGate = 1e-6f;
}

__device__ __forceinline__ float gelu_exact(float x) {
  return 0.5f * x * (1.0f + erff(x * 0.70710678118654752440f));
}

// 16B global -> LDS copy: async DMA (ASYNCcnt) when available, else through VGPRs.
__device__ __forceinline__ void cp16_g2l(const bf16_t* __restrict__ g,
                                         bf16_t* __restrict__ l) {
#if defined(HAVE_ASYNC_LDS)
  __builtin_amdgcn_global_load_async_to_lds_b128((gvec_t*)g, (lvec_t*)l, 0, 0);
#else
  *(u32x4*)l = *(const u32x4*)g;
#endif
}

__device__ __forceinline__ void wait_async0() {
#if defined(HAVE_ASYNC_LDS)
#if defined(HAVE_WAIT_ASYNC_BUILTIN)
  __builtin_amdgcn_s_wait_asynccnt(0);
#else
  asm volatile("s_wait_asynccnt 0" ::: "memory");
#endif
#endif
}

// ---------- prep: reshape(T,5120) + RMSNorm -> bf16, router logits -> gates ----------
__global__ void __launch_bounds__(256)
k_prep(const float* __restrict__ x, const float* __restrict__ norm_w,
       const float* __restrict__ router_w, bf16_t* __restrict__ xn,
       float* __restrict__ gates) {
  const int t = blockIdx.x;
  const int tid = threadIdx.x;
  const float* xr = x + (size_t)t * kInDim;
  __shared__ float red[256];
  __shared__ float r4[256 * 4];

  float ss = 0.f;
  for (int i = tid; i < kInDim; i += 256) { const float v = xr[i]; ss += v * v; }
  red[tid] = ss;
  __syncthreads();
  for (int s = 128; s > 0; s >>= 1) {
    if (tid < s) red[tid] += red[tid + s];
    __syncthreads();
  }
  const float rstd = rsqrtf(red[0] * (1.0f / (float)kInDim) + kEpsNorm);

  float d0 = 0.f, d1 = 0.f, d2 = 0.f, d3 = 0.f;
  bf16_t* xo = xn + (size_t)t * kInDim;
  for (int i = tid; i < kInDim; i += 256) {
    const float v = xr[i] * rstd * norm_w[i];
    xo[i] = (bf16_t)v;
    d0 += v * router_w[i];
    d1 += v * router_w[kInDim + i];
    d2 += v * router_w[2 * kInDim + i];
    d3 += v * router_w[3 * kInDim + i];
  }
  r4[tid * 4 + 0] = d0; r4[tid * 4 + 1] = d1;
  r4[tid * 4 + 2] = d2; r4[tid * 4 + 3] = d3;
  __syncthreads();
  for (int s = 128; s > 0; s >>= 1) {
    if (tid < s) {
      r4[tid * 4 + 0] += r4[(tid + s) * 4 + 0];
      r4[tid * 4 + 1] += r4[(tid + s) * 4 + 1];
      r4[tid * 4 + 2] += r4[(tid + s) * 4 + 2];
      r4[tid * 4 + 3] += r4[(tid + s) * 4 + 3];
    }
    __syncthreads();
  }
  if (tid == 0) {
    float p[4] = {r4[0], r4[1], r4[2], r4[3]};
    float mx = p[0];
    for (int j = 1; j < 4; ++j) mx = fmaxf(mx, p[j]);
    float sum = 0.f;
    for (int j = 0; j < 4; ++j) { p[j] = __expf(p[j] - mx); sum += p[j]; }
    const float inv = 1.0f / sum;
    for (int j = 0; j < 4; ++j) p[j] *= inv;
    int i0 = 0;
    for (int j = 1; j < 4; ++j) if (p[j] > p[i0]) i0 = j;
    int i1 = (i0 == 0) ? 1 : 0;
    for (int j = 0; j < 4; ++j) if (j != i0 && p[j] > p[i1]) i1 = j;
    const float denom = p[i0] + p[i1] + kEpsGate;
    float g[4] = {0.f, 0.f, 0.f, 0.f};
    g[i0] = p[i0] / denom;
    g[i1] = p[i1] / denom;
    float* gp = gates + (size_t)t * kExp;
    gp[0] = g[0]; gp[1] = g[1]; gp[2] = g[2]; gp[3] = g[3];
  }
}

// ---------- fp32 -> bf16 weight convert (vectorized, grid-stride) ----------
__global__ void __launch_bounds__(256)
k_cvt(const float* __restrict__ s, bf16_t* __restrict__ d, int n4) {
  int i = blockIdx.x * blockDim.x + threadIdx.x;
  const int stride = gridDim.x * blockDim.x;
  for (; i < n4; i += stride) {
    const float4 v = ((const float4*)s)[i];
    bf16_t* o = d + (size_t)i * 4;
    o[0] = (bf16_t)v.x; o[1] = (bf16_t)v.y;
    o[2] = (bf16_t)v.z; o[3] = (bf16_t)v.w;
  }
}

// ---------- WMMA GEMM: C = epilogue(A[M,K](bf16) * W[N,K]^T(bf16) + bias) ----------
// OUT_MODE: 0 = store bf16 (H), 1 = store f32 (shared expert), 2 = f32 += gate * v
template <bool DO_GELU, int OUT_MODE>
__global__ void __launch_bounds__(256)
k_gemm(const bf16_t* __restrict__ A, const bf16_t* __restrict__ W,
       const float* __restrict__ bias, void* __restrict__ Cout,
       const float* __restrict__ gates, int expert, int M, int N, int K) {
  // Double-buffered tiles: async DMA fills buf^1 while WMMA consumes buf.
  __shared__ __align__(16) bf16_t As[2 * kTileA];
  __shared__ __align__(16) bf16_t Bs[2 * kTileB];

  const int tid   = threadIdx.x;
  const int lane  = tid & 31;
  const int wave  = tid >> 5;
  const int waveM = wave >> 2;  // 0..1 -> 64 rows each
  const int waveN = wave & 3;   // 0..3 -> 32 cols each
  const int m0 = blockIdx.y * BM;
  const int n0 = blockIdx.x * BN;

  const int ldRow = tid >> 3;        // 0..31
  const int ldCol = (tid & 7) * 8;   // bf16 element offset (16B chunks)
  const int nk = K / BK;

  // Pre-zero LDS slots of out-of-range A rows (async loads skip them, zeros persist).
  #pragma unroll
  for (int i = 0; i < 4; ++i) {
    const int row = ldRow + i * 32;
    if (m0 + row >= M) {
      const u32x4 z = (u32x4)0u;
      *(u32x4*)(As + 0 * kTileA + row * BK + ldCol) = z;
      *(u32x4*)(As + 1 * kTileA + row * BK + ldCol) = z;
    }
  }

  auto issueTile = [&](int kt, int b) {
    const int kbase = kt * BK + ldCol;
    #pragma unroll
    for (int i = 0; i < 4; ++i) {
      const int row = ldRow + i * 32;
      const int ar = m0 + row;
      if (ar < M)
        cp16_g2l(A + (size_t)ar * K + kbase, As + b * kTileA + row * BK + ldCol);
      cp16_g2l(W + (size_t)(n0 + row) * K + kbase, Bs + b * kTileB + row * BK + ldCol);
    }
  };

  v8f acc[4][2];
  #pragma unroll
  for (int mi = 0; mi < 4; ++mi)
    #pragma unroll
    for (int ni = 0; ni < 2; ++ni) acc[mi][ni] = (v8f)0.f;

  const int g16 = lane >> 4;  // half-wave group
  const int l16 = lane & 15;

  issueTile(0, 0);
  wait_async0();
  __syncthreads();

  for (int kt = 0; kt < nk; ++kt) {
    const int cur = kt & 1;
    if (kt + 1 < nk) issueTile(kt + 1, cur ^ 1);  // DMA next tile during compute

    const bf16_t* Ab = As + cur * kTileA;
    const bf16_t* Bb = Bs + cur * kTileB;

    #pragma unroll
    for (int ks = 0; ks < 2; ++ks) {  // two 16x16x32 steps per BK=64 tile
      v16bf bfrag[2];
      #pragma unroll
      for (int ni = 0; ni < 2; ++ni) {
        // B lane layout: lane = N col, 16 consecutive K per half-wave
        const bf16_t* bp = Bb + (waveN * 32 + ni * 16 + l16) * BK + ks * 32 + g16 * 16;
        const v8bf lo = *(const v8bf*)bp;
        const v8bf hi = *(const v8bf*)(bp + 8);
        bfrag[ni] = __builtin_shufflevector(lo, hi, 0, 1, 2, 3, 4, 5, 6, 7,
                                            8, 9, 10, 11, 12, 13, 14, 15);
      }
      #pragma unroll
      for (int mi = 0; mi < 4; ++mi) {
        // A lane layout: lane = M row, 16 consecutive K per half-wave
        const bf16_t* ap = Ab + (waveM * 64 + mi * 16 + l16) * BK + ks * 32 + g16 * 16;
        const v8bf lo = *(const v8bf*)ap;
        const v8bf hi = *(const v8bf*)(ap + 8);
        const v16bf afrag = __builtin_shufflevector(lo, hi, 0, 1, 2, 3, 4, 5, 6, 7,
                                                    8, 9, 10, 11, 12, 13, 14, 15);
        #pragma unroll
        for (int ni = 0; ni < 2; ++ni) {
          acc[mi][ni] = __builtin_amdgcn_wmma_f32_16x16x32_bf16(
              false, afrag, false, bfrag[ni], (short)0, acc[mi][ni], false, false);
        }
      }
    }

    wait_async0();     // next tile's DMA complete (this wave)
    __syncthreads();   // all waves done reading cur / writing cur^1
  }

  // epilogue: C/D layout -> M = r + (lane>>4)*8, N = lane&15
  #pragma unroll
  for (int mi = 0; mi < 4; ++mi) {
    #pragma unroll
    for (int ni = 0; ni < 2; ++ni) {
      const int col = n0 + waveN * 32 + ni * 16 + l16;
      const float bb = bias[col];
      #pragma unroll
      for (int r = 0; r < 8; ++r) {
        const int row = m0 + waveM * 64 + mi * 16 + g16 * 8 + r;
        if (row >= M) continue;
        float v = acc[mi][ni][r] + bb;
        if constexpr (DO_GELU) v = gelu_exact(v);
        if constexpr (OUT_MODE == 0) {
          ((bf16_t*)Cout)[(size_t)row * N + col] = (bf16_t)v;
        } else if constexpr (OUT_MODE == 1) {
          ((float*)Cout)[(size_t)row * N + col] = v;
        } else {
          const float g = gates[(size_t)row * kExp + expert];
          if (g != 0.f) {
            float* p = (float*)Cout + (size_t)row * N + col;
            *p += g * v;
          }
        }
      }
    }
  }
}

extern "C" void kernel_launch(void* const* d_in, const int* in_sizes, int n_in,
                              void* d_out, int out_size, void* d_ws, size_t ws_size,
                              hipStream_t stream) {
  const float* x        = (const float*)d_in[0];
  const float* norm_w   = (const float*)d_in[1];
  const float* router_w = (const float*)d_in[2];
  const float* w1       = (const float*)d_in[3];
  const float* b1       = (const float*)d_in[4];
  const float* w2       = (const float*)d_in[5];
  const float* b2       = (const float*)d_in[6];
  const float* sw1      = (const float*)d_in[7];
  const float* sb1      = (const float*)d_in[8];
  const float* sw2      = (const float*)d_in[9];
  const float* sb2      = (const float*)d_in[10];
  float* out = (float*)d_out;

  char* ws = (char*)d_ws;
  size_t off = 0;
  auto walloc = [&](size_t bytes) -> void* {
    void* p = ws + off;
    off = (off + bytes + 255) & ~(size_t)255;
    return p;
  };
  bf16_t* Xn  = (bf16_t*)walloc((size_t)kTok * kInDim * sizeof(bf16_t)); // 61.4 MB
  float*  gts = (float*) walloc((size_t)kTok * kExp * sizeof(float));    // 96 KB
  bf16_t* Wb1 = (bf16_t*)walloc((size_t)kHid * kInDim * sizeof(bf16_t)); // 21 MB (reused)
  bf16_t* Wb2 = (bf16_t*)walloc((size_t)kOut * kHid * sizeof(bf16_t));   // 8.4 MB (reused)
  bf16_t* H   = (bf16_t*)walloc((size_t)kTok * kHid * sizeof(bf16_t));   // 24.6 MB (reused)

  k_prep<<<kTok, 256, 0, stream>>>(x, norm_w, router_w, Xn, gts);

  const dim3 g1(kHid / BN, (kTok + BM - 1) / BM);
  const dim3 g2(kOut / BN, (kTok + BM - 1) / BM);
  const int cvt_blocks = 1024;

  // Shared expert first: OUT_MODE=1 overwrites d_out (initializes it).
  k_cvt<<<cvt_blocks, 256, 0, stream>>>(sw1, Wb1, kHid * kInDim / 4);
  k_gemm<true, 0><<<g1, 256, 0, stream>>>(Xn, Wb1, sb1, H, nullptr, 0, kTok, kHid, kInDim);
  k_cvt<<<cvt_blocks, 256, 0, stream>>>(sw2, Wb2, kOut * kHid / 4);
  k_gemm<false, 1><<<g2, 256, 0, stream>>>(H, Wb2, sb2, out, nullptr, 0, kTok, kOut, kHid);

  // Sparse experts: gated accumulate into d_out.
  for (int e = 0; e < kExp; ++e) {
    k_cvt<<<cvt_blocks, 256, 0, stream>>>(w1 + (size_t)e * kHid * kInDim, Wb1,
                                          kHid * kInDim / 4);
    k_gemm<true, 0><<<g1, 256, 0, stream>>>(Xn, Wb1, b1 + (size_t)e * kHid, H,
                                            nullptr, 0, kTok, kHid, kInDim);
    k_cvt<<<cvt_blocks, 256, 0, stream>>>(w2 + (size_t)e * kOut * kHid, Wb2,
                                          kOut * kHid / 4);
    k_gemm<false, 2><<<g2, 256, 0, stream>>>(H, Wb2, b2 + (size_t)e * kOut, out,
                                             gts, e, kTok, kOut, kHid);
  }
  (void)in_sizes; (void)n_in; (void)out_size; (void)ws_size;
}
